// MultiHeadedAttention_40724879900854
// MI455X (gfx1250) — compile-verified, matrix operand order checked by reading, not run on hardware
//
#include <hip/hip_runtime.h>
#include <hip/hip_bf16.h>
#include <stdint.h>

// MI455X / gfx1250 (CDNA5, wave32). Linear-attention rewrite:
//   out = Q' (K'^T V') + resid   (exact: no softmax => associativity).
// bf16 WMMA w/ f32 accum everywhere; W-operand tiles staged into LDS by the
// Tensor Data Mover (double-buffered, s_wait_tensorcnt), A streamed from
// global with prefetch hints.

typedef __bf16 bf16;
typedef __attribute__((ext_vector_type(16))) __bf16 v16bf;
typedef __attribute__((ext_vector_type(8)))  __bf16 v8bf;
typedef __attribute__((ext_vector_type(4)))  __bf16 v4bf;
typedef __attribute__((ext_vector_type(8)))  float  v8f;
typedef __attribute__((ext_vector_type(2)))  float  v2f;

#define BATCH 2
#define S_LEN 2048
#define DM    1024
#define NH    16
#define DK    64

#define LDSH  40            // padded LDS row pitch (elements): 32 data + 4 DW pad

#ifndef __has_builtin
#define __has_builtin(x) 0
#endif
#if __has_builtin(__builtin_amdgcn_tensor_load_to_lds)
#define HAVE_TDM 1
#else
#define HAVE_TDM 0
#endif

// ---------------------------------------------------------------------------
// fp32 -> bf16 conversion (memory-bound; halves all GEMM traffic)
// ---------------------------------------------------------------------------
__global__ __launch_bounds__(256)
void cvt_f32_to_bf16(const float* __restrict__ in, bf16* __restrict__ out, int n4) {
    int i = blockIdx.x * blockDim.x + threadIdx.x;
    if (i >= n4) return;
    const float4 f = *(const float4*)(in + (size_t)i * 4);
    v4bf o;
    o[0] = (bf16)f.x; o[1] = (bf16)f.y; o[2] = (bf16)f.z; o[3] = (bf16)f.w;
    *(v4bf*)(out + (size_t)i * 4) = o;
}

// ---------------------------------------------------------------------------
// bf16 fragment load, ISA 16-bit A/B layout (16x32 MxK striping):
//   lane<16 : K = {k0..k0+7, k0+16..k0+23}
//   lane>=16: K = {k0+8..k0+15, k0+24..k0+31}
// ---------------------------------------------------------------------------
union FragU { v16bf v; v8bf h[2]; };

__device__ __forceinline__ v16bf load_frag(const bf16* base, int ld,
                                           int row0, int k0, int lane) {
    const bf16* p = base + (size_t)(row0 + (lane & 15)) * ld + k0 + ((lane >> 4) << 3);
    FragU u;
    u.h[0] = *(const v8bf*)(p);
    u.h[1] = *(const v8bf*)(p + 16);
    return u.v;
}

// ---------------------------------------------------------------------------
// TDM: DMA a 128(n) x 32(k) bf16 tile of W into LDS, padded to 80B row pitch
// (pad 4 DWORDs after every 16 DWORDs) => 16B-aligned, conflict-free rows.
// D# layout per CDNA5 ISA.
// ---------------------------------------------------------------------------
#if HAVE_TDM
typedef __attribute__((ext_vector_type(4))) unsigned int u32x4;
typedef __attribute__((ext_vector_type(8))) int          i32x8;
typedef __attribute__((ext_vector_type(4))) int          i32x4;

__device__ __forceinline__ void tdm_issue_w(const bf16* gsrc, unsigned lds_off,
                                            int K, int N) {
    unsigned long long ga = (unsigned long long)(uintptr_t)gsrc;
    u32x4 g0;
    g0[0] = 1u;                                   // count=1, user mode, no gather
    g0[1] = lds_off;                              // lds_addr (bytes)
    g0[2] = (unsigned)ga;                         // global_addr[31:0]
    g0[3] = (unsigned)((ga >> 32) & 0x01FFFFFFull) | (2u << 30);  // addr[56:32] | type=2
    i32x8 g1;
    // data_size=2B(1)<<16 | pad_enable<<20 | pad_interval=16DW(3)<<22 | pad_amount=4DW(3)<<25
    g1[0] = (1 << 16) | (1 << 20) | (3 << 22) | (3 << 25);
    g1[1] = (K & 0xFFFF) << 16;                   // tensor_dim0[15:0]
    g1[2] = ((K >> 16) & 0xFFFF) | ((N & 0xFFFF) << 16);  // td0 hi | tensor_dim1 lo
    g1[3] = ((N >> 16) & 0xFFFF) | (32 << 16);    // td1 hi | tile_dim0 = 32
    g1[4] = 128;                                  // tile_dim1 = 128, tile_dim2 = 0
    g1[5] = K;                                    // tensor_dim0_stride[31:0]
    g1[6] = 0;                                    // stride hi | dim1_stride lo (unused, 2D)
    g1[7] = 0;
    i32x4 gz = {0, 0, 0, 0};
#if __clang_major__ >= 23
    i32x8 gz8 = {0, 0, 0, 0, 0, 0, 0, 0};
    __builtin_amdgcn_tensor_load_to_lds(g0, g1, gz, gz, gz8, 0);
#else
    __builtin_amdgcn_tensor_load_to_lds(g0, g1, gz, gz, 0);
#endif
}
#endif

__device__ __forceinline__ void fill_w_tile(bf16* dst, const bf16* W,
                                            int n_blk, int k0, int K, int tid) {
#if HAVE_TDM
    if (tid < 32)                                 // one wave issues one TDM op
        tdm_issue_w(W + (size_t)n_blk * K + k0, (unsigned)(uintptr_t)dst, K, DM);
#else
    for (int t = tid; t < (128 * 32) / 8; t += 128) {
        int r = t >> 2, c = (t & 3) * 8;
        *(v8bf*)(dst + r * LDSH + c) = *(const v8bf*)(W + (size_t)(n_blk + r) * K + k0 + c);
    }
#endif
}

__device__ __forceinline__ void wait_w_tile(int tid) {
#if HAVE_TDM
    if (tid < 32) __builtin_amdgcn_s_wait_tensorcnt(0);
#endif
}

// ---------------------------------------------------------------------------
// Projection GEMM: C[M,N](bf16) = A[M,K](bf16) * W[N,K]^T(bf16), f32 accum.
// Block: 128 threads = 4 waves (2x2); wave tile 32 rows x 64 cols.
// W tile (128n x 32k) double-buffered in LDS via TDM; A direct + prefetch.
// ---------------------------------------------------------------------------
__global__ __launch_bounds__(128)
void gemm_proj(const bf16* __restrict__ A, const bf16* __restrict__ W,
               bf16* __restrict__ C, int M, int N, int K) {
    __shared__ bf16 smW[2][128 * LDSH];
    const int tid  = threadIdx.x;
    const int lane = tid & 31;
    const int wave = tid >> 5;                    // 0..3
    const int m0   = blockIdx.y * 64 + (wave >> 1) * 32;
    const int n_blk = blockIdx.x * 128;
    const int wn64  = (wave & 1) * 64;

    fill_w_tile(smW[0], W, n_blk, 0, K, tid);     // prologue -> buf 0

    v8f acc[2][4] = {};
    int buf = 0;
    for (int k0 = 0; k0 < K; k0 += 32) {
        wait_w_tile(tid);                         // s_wait_tensorcnt 0 (wave 0)
        __syncthreads();                          // publish smW[buf]
        if (k0 + 32 < K)
            fill_w_tile(smW[buf ^ 1], W, n_blk, k0 + 32, K, tid);   // overlap DMA

        __builtin_prefetch(A + (size_t)(m0 + lane % 16) * K + k0 + 256, 0, 1);

        v16bf a0 = load_frag(A, K, m0,      k0, lane);
        v16bf a1 = load_frag(A, K, m0 + 16, k0, lane);
        v16bf b[4];
#pragma unroll
        for (int j = 0; j < 4; ++j)
            b[j] = load_frag(smW[buf], LDSH, wn64 + j * 16, 0, lane);  // ds_load
#pragma unroll
        for (int j = 0; j < 4; ++j) {
            acc[0][j] = __builtin_amdgcn_wmma_f32_16x16x32_bf16(
                false, a0, false, b[j], (short)0, acc[0][j], false, false);
            acc[1][j] = __builtin_amdgcn_wmma_f32_16x16x32_bf16(
                false, a1, false, b[j], (short)0, acc[1][j], false, false);
        }
        __syncthreads();                          // everyone done with smW[buf]
        buf ^= 1;
    }

    const int lm   = lane & 15;
    const int rsel = (lane >> 4) << 3;            // C/D layout: lanes>=16 hold M=r+8
#pragma unroll
    for (int i = 0; i < 2; ++i)
#pragma unroll
        for (int j = 0; j < 4; ++j)
#pragma unroll
            for (int r = 0; r < 8; ++r) {
                int row = m0 + i * 16 + r + rsel;
                int col = n_blk + wn64 + j * 16 + lm;
                C[(size_t)row * N + col] = (bf16)acc[i][j][r];
            }
}

// ---------------------------------------------------------------------------
// Stage 2: M = Kh^T * Vh per (b,h): [64x64] = [64x2048]*[2048x64].
// v_wmma_f32_16x16x4_f32: f32 A-layout puts lanes along d => no transpose
// gather. 8 waves split s; cross-wave reduce via ds_add_f32. Emits M^T bf16.
// ---------------------------------------------------------------------------
__global__ __launch_bounds__(256)
void ktv_small(const bf16* __restrict__ Kb, const bf16* __restrict__ Vb,
               bf16* __restrict__ MT) {
    __shared__ float Ms[64 * 64];
    const int bh = blockIdx.x;                    // b*16 + h
    const int b  = bh >> 4, h = bh & 15;
    const int col0 = h * DK;
    const size_t rowbase = (size_t)b * S_LEN * DM;

    for (int i = threadIdx.x; i < 64 * 64; i += 256) Ms[i] = 0.f;
    __syncthreads();

    const int lane = threadIdx.x & 31;
    const int wave = threadIdx.x >> 5;            // 0..7
    const int lm   = lane & 15;
    const int klo  = (lane >> 4) * 2;             // f32 WMMA: lanes>=16 hold K=2,3
    const int s_begin = wave * (S_LEN / 8);
    const int s_end   = s_begin + (S_LEN / 8);

    v8f acc[4][4] = {};
    for (int s = s_begin; s < s_end; s += 4) {
        v2f a[4], bb[4];
#pragma unroll
        for (int i = 0; i < 4; ++i) {
            const bf16* pk = Kb + rowbase + (size_t)(s + klo) * DM + col0 + i * 16 + lm;
            const bf16* pv = Vb + rowbase + (size_t)(s + klo) * DM + col0 + i * 16 + lm;
            a[i][0]  = (float)pk[0];  a[i][1]  = (float)pk[DM];
            bb[i][0] = (float)pv[0];  bb[i][1] = (float)pv[DM];
        }
#pragma unroll
        for (int i = 0; i < 4; ++i)
#pragma unroll
            for (int j = 0; j < 4; ++j)
                acc[i][j] = __builtin_amdgcn_wmma_f32_16x16x4_f32(
                    false, a[i], false, bb[j], (short)0, acc[i][j], false, false);
    }

    const int rsel = (lane >> 4) << 3;
#pragma unroll
    for (int i = 0; i < 4; ++i)
#pragma unroll
        for (int j = 0; j < 4; ++j)
#pragma unroll
            for (int r = 0; r < 8; ++r) {
                int d = i * 16 + r + rsel;
                int e = j * 16 + lm;
                atomicAdd(&Ms[d * 64 + e], acc[i][j][r]);   // ds_add_f32
            }
    __syncthreads();

    bf16* outp = MT + (size_t)bh * 64 * 64;       // MT[e][d]
    for (int idx = threadIdx.x; idx < 64 * 64; idx += 256) {
        int e = idx >> 6, d = idx & 63;
        outp[idx] = (bf16)Ms[d * 64 + e];
    }
}

// ---------------------------------------------------------------------------
// Stage 3: x = Qh * M, fused residual add, fp32 output.
// Block: 256 threads = 8 waves; wave tile 32 s-rows x 64 e-cols, K=64.
// ---------------------------------------------------------------------------
__global__ __launch_bounds__(256)
void qm_residual(const bf16* __restrict__ Qb, const bf16* __restrict__ MT,
                 const float* __restrict__ resid, float* __restrict__ out) {
    const int bh = blockIdx.y;
    const int b  = bh >> 4, h = bh & 15;
    const int lane = threadIdx.x & 31;
    const int wave = threadIdx.x >> 5;
    const int s0 = blockIdx.x * 256 + wave * 32;
    const size_t rowbase = (size_t)b * S_LEN * DM;
    const bf16* Mh = MT + (size_t)bh * 64 * 64;
    const bf16* Ah = Qb + rowbase + h * DK;

    v8f acc[2][4] = {};
#pragma unroll
    for (int k0 = 0; k0 < 64; k0 += 32) {
        v16bf a0 = load_frag(Ah, DM, s0,      k0, lane);
        v16bf a1 = load_frag(Ah, DM, s0 + 16, k0, lane);
        v16bf bb[4];
#pragma unroll
        for (int j = 0; j < 4; ++j) bb[j] = load_frag(Mh, 64, j * 16, k0, lane);
#pragma unroll
        for (int j = 0; j < 4; ++j) {
            acc[0][j] = __builtin_amdgcn_wmma_f32_16x16x32_bf16(
                false, a0, false, bb[j], (short)0, acc[0][j], false, false);
            acc[1][j] = __builtin_amdgcn_wmma_f32_16x16x32_bf16(
                false, a1, false, bb[j], (short)0, acc[1][j], false, false);
        }
    }

    const int lm   = lane & 15;
    const int rsel = (lane >> 4) << 3;
#pragma unroll
    for (int i = 0; i < 2; ++i)
#pragma unroll
        for (int j = 0; j < 4; ++j)
#pragma unroll
            for (int r = 0; r < 8; ++r) {
                int row = s0 + i * 16 + r + rsel;
                int col = h * DK + j * 16 + lm;
                size_t idx = rowbase + (size_t)row * DM + col;
                out[idx] = resid[idx] + acc[i][j][r];
            }
}

// ---------------------------------------------------------------------------
// Host launcher
// ---------------------------------------------------------------------------
extern "C" void kernel_launch(void* const* d_in, const int* in_sizes, int n_in,
                              void* d_out, int out_size, void* d_ws, size_t ws_size,
                              hipStream_t stream) {
    const float* q  = (const float*)d_in[0];
    const float* k  = (const float*)d_in[1];
    const float* v  = (const float*)d_in[2];
    const float* W1 = (const float*)d_in[3];
    const float* W2 = (const float*)d_in[4];
    const float* W3 = (const float*)d_in[5];
    float* out = (float*)d_out;

    const size_t NQKV = (size_t)BATCH * S_LEN * DM;   // 4M elements
    const size_t NW   = (size_t)DM * DM;              // 1M elements

    bf16* ws = (bf16*)d_ws;
    bf16* xq  = ws;                  // converted inputs (bf16)
    bf16* xk  = xq  + NQKV;
    bf16* xv  = xk  + NQKV;
    bf16* w1b = xv  + NQKV;
    bf16* w2b = w1b + NW;
    bf16* w3b = w2b + NW;
    bf16* qb  = w3b + NW;            // projected Q/K/V (bf16)
    bf16* kb  = qb  + NQKV;
    bf16* vb  = kb  + NQKV;
    bf16* MT  = vb  + NQKV;          // per-head (K^T V)^T, 32 * 64 * 64

    // 1) fp32 -> bf16 conversions
    cvt_f32_to_bf16<<<dim3((unsigned)(NQKV / 4 / 256)), dim3(256), 0, stream>>>(q,  xq, (int)(NQKV / 4));
    cvt_f32_to_bf16<<<dim3((unsigned)(NQKV / 4 / 256)), dim3(256), 0, stream>>>(k,  xk, (int)(NQKV / 4));
    cvt_f32_to_bf16<<<dim3((unsigned)(NQKV / 4 / 256)), dim3(256), 0, stream>>>(v,  xv, (int)(NQKV / 4));
    cvt_f32_to_bf16<<<dim3((unsigned)(NW   / 4 / 256)), dim3(256), 0, stream>>>(W1, w1b, (int)(NW / 4));
    cvt_f32_to_bf16<<<dim3((unsigned)(NW   / 4 / 256)), dim3(256), 0, stream>>>(W2, w2b, (int)(NW / 4));
    cvt_f32_to_bf16<<<dim3((unsigned)(NW   / 4 / 256)), dim3(256), 0, stream>>>(W3, w3b, (int)(NW / 4));

    // 2) projections: [4096,1024] = [4096,1024] x [1024,1024]^T
    const dim3 ggrid(DM / 128, (BATCH * S_LEN) / 64);
    gemm_proj<<<ggrid, dim3(128), 0, stream>>>(xq, w1b, qb, BATCH * S_LEN, DM, DM);
    gemm_proj<<<ggrid, dim3(128), 0, stream>>>(xk, w2b, kb, BATCH * S_LEN, DM, DM);
    gemm_proj<<<ggrid, dim3(128), 0, stream>>>(xv, w3b, vb, BATCH * S_LEN, DM, DM);

    // 3) per-head M^T = (K^T V)^T
    ktv_small<<<dim3(BATCH * NH), dim3(256), 0, stream>>>(kb, vb, MT);

    // 4) x = Q*M + residual
    qm_residual<<<dim3(S_LEN / 256, BATCH * NH), dim3(256), 0, stream>>>(qb, MT, q, out);
}
